// BAKT_32006096290477
// MI455X (gfx1250) — matrix operand order repaired and checked by reading, not compiled, not run on hardware
//
#include <hip/hip_runtime.h>

#define B_  32
#define S_  512
#define D_  512
#define H_  8
#define L_  4
#define F_  2048
#define DK_ 64
#define EPS_ 1e-5f

typedef _Float16 half_t;
typedef __attribute__((ext_vector_type(16))) _Float16 v16h;
typedef __attribute__((ext_vector_type(8)))  _Float16 v8h;
typedef __attribute__((ext_vector_type(8)))  float    v8f;

#define WMMA_F16(a, b, c) \
    __builtin_amdgcn_wmma_f32_16x16x32_f16(false, (a), false, (b), (short)0, (c), false, false)

// ---------------------------------------------------------------------------
// Fragment helpers (WMMA 16x16x32 f16 operands; contiguous k order chosen here, same for A and B)
// A 16x32: lane L<16 -> row L, K = 0..15 ; lane L>=16 -> row L-16, K = 16..31
// B 32x16: lane L<16 -> col L, K = 0..15 ; lane L>=16 -> col L-16, K = 16..31
// ---------------------------------------------------------------------------
__device__ __forceinline__ v16h frag_a_from(const half_t* rowptr, int hl) {
    return *(const v16h*)(rowptr + hl * 16);
}

// 16-lane XOR-butterfly max via DPP16 (stays inside each 16-lane half-wave;
// wave32 has exactly two DPP rows, matching our two row-groups).
// DPP control must be an immediate -> template parameter.
template<int CTRL>
__device__ __forceinline__ float fmax_dpp(float x) {
    int m = __builtin_amdgcn_update_dpp(0, __float_as_int(x), CTRL, 0xF, 0xF, true);
    return fmaxf(x, __int_as_float(m));
}
__device__ __forceinline__ float rowmax16(float x) {
    x = fmax_dpp<0xB1>(x);    // xor 1 : quad_perm(1,0,3,2)
    x = fmax_dpp<0x4E>(x);    // xor 2 : quad_perm(2,3,0,1)
    x = fmax_dpp<0x141>(x);   // xor 4 : row_half_mirror
    x = fmax_dpp<0x140>(x);   // xor 8 : row_mirror
    return x;
}

// ---------------------------------------------------------------------------
// f32 -> f16 convert
// ---------------------------------------------------------------------------
__global__ void cvt_f16_kernel(const float* __restrict__ src,
                               half_t* __restrict__ dst, int n) {
    int i = blockIdx.x * 256 + threadIdx.x;
    if (i < n) dst[i] = (half_t)src[i];
}

// ---------------------------------------------------------------------------
// x = q + pos ; y = qa + pos  (f32 x into d_out, f16 shadows)
// ---------------------------------------------------------------------------
__global__ void add_pos_kernel(const float* __restrict__ q,
                               const float* __restrict__ qa,
                               const float* __restrict__ pos,
                               float* __restrict__ x,
                               half_t* __restrict__ xh,
                               half_t* __restrict__ yh) {
    size_t i  = (size_t)blockIdx.x * 256 + threadIdx.x;
    size_t sd = i % ((size_t)S_ * D_);
    float p  = pos[sd];
    float xv = q[i] + p;
    float yv = qa[i] + p;
    x[i]  = xv;
    xh[i] = (half_t)xv;
    yh[i] = (half_t)yv;
}

// ---------------------------------------------------------------------------
// GEMM: C[M,N] = A[M,K](f16) @ W[N,K](f16)^T + bias[N]
// 256 threads = 8 waves; wave -> 32x64 C (8 wmma tiles); block -> 128x128 C.
// Software-pipelined K loop.
// ---------------------------------------------------------------------------
template<bool RELU, bool F16OUT>
__global__ __launch_bounds__(256)
void gemm_wmma_kernel(const half_t* __restrict__ A,
                      const half_t* __restrict__ W,
                      const float*  __restrict__ bias,
                      half_t* __restrict__ Cf16,
                      float*  __restrict__ Cf32,
                      int M, int N, int K) {
    const int lane = threadIdx.x & 31;
    const int wave = threadIdx.x >> 5;
    const int m0 = blockIdx.x * 128 + (wave >> 1) * 32;
    const int n0 = blockIdx.y * 128 + (wave & 1) * 64;
    const int row = lane & 15;
    const int hl  = lane >> 4;

    const half_t* Ar0 = A + (size_t)(m0 + row) * K;
    const half_t* Ar1 = A + (size_t)(m0 + 16 + row) * K;
    const half_t* Wr0 = W + (size_t)(n0 + row) * K;
    const half_t* Wr1 = Wr0 + (size_t)16 * K;
    const half_t* Wr2 = Wr0 + (size_t)32 * K;
    const half_t* Wr3 = Wr0 + (size_t)48 * K;

    v8f acc[8];
    #pragma unroll
    for (int i = 0; i < 8; ++i) acc[i] = (v8f){};

    v16h a0 = frag_a_from(Ar0, hl);
    v16h a1 = frag_a_from(Ar1, hl);
    v16h b0 = *(const v16h*)(Wr0 + hl * 16);
    v16h b1 = *(const v16h*)(Wr1 + hl * 16);
    v16h b2 = *(const v16h*)(Wr2 + hl * 16);
    v16h b3 = *(const v16h*)(Wr3 + hl * 16);

    const int kiter = K >> 5;
    for (int it = 0; it < kiter - 1; ++it) {
        const int kn = (it + 1) << 5;
        v16h na0 = frag_a_from(Ar0 + kn, hl);
        v16h na1 = frag_a_from(Ar1 + kn, hl);
        v16h nb0 = *(const v16h*)(Wr0 + kn + hl * 16);
        v16h nb1 = *(const v16h*)(Wr1 + kn + hl * 16);
        v16h nb2 = *(const v16h*)(Wr2 + kn + hl * 16);
        v16h nb3 = *(const v16h*)(Wr3 + kn + hl * 16);
        acc[0] = WMMA_F16(a0, b0, acc[0]);
        acc[1] = WMMA_F16(a0, b1, acc[1]);
        acc[2] = WMMA_F16(a0, b2, acc[2]);
        acc[3] = WMMA_F16(a0, b3, acc[3]);
        acc[4] = WMMA_F16(a1, b0, acc[4]);
        acc[5] = WMMA_F16(a1, b1, acc[5]);
        acc[6] = WMMA_F16(a1, b2, acc[6]);
        acc[7] = WMMA_F16(a1, b3, acc[7]);
        a0 = na0; a1 = na1;
        b0 = nb0; b1 = nb1; b2 = nb2; b3 = nb3;
    }
    acc[0] = WMMA_F16(a0, b0, acc[0]);
    acc[1] = WMMA_F16(a0, b1, acc[1]);
    acc[2] = WMMA_F16(a0, b2, acc[2]);
    acc[3] = WMMA_F16(a0, b3, acc[3]);
    acc[4] = WMMA_F16(a1, b0, acc[4]);
    acc[5] = WMMA_F16(a1, b1, acc[5]);
    acc[6] = WMMA_F16(a1, b2, acc[6]);
    acc[7] = WMMA_F16(a1, b3, acc[7]);

    #pragma unroll
    for (int g = 0; g < 2; ++g) {
        #pragma unroll
        for (int t = 0; t < 4; ++t) {
            int n = n0 + t * 16 + row;
            float bn = bias[n];
            #pragma unroll
            for (int j = 0; j < 8; ++j) {
                int m = m0 + g * 16 + j + 8 * hl;
                float vv = acc[g * 4 + t][j] + bn;
                if (RELU) vv = fmaxf(vv, 0.0f);
                if (F16OUT) Cf16[(size_t)m * N + n] = (half_t)vv;
                else        Cf32[(size_t)m * N + n] = vv;
            }
        }
    }
}

// ---------------------------------------------------------------------------
// Flash attention per (b,h): scores = (kq kq^T) * scale * fr[s], strict-lower
// causal mask (-1e32), online softmax, row 0 -> 0, ctx = attn @ v.
// Row max: DPP butterfly (pure VALU). Row sum: P @ ones via an extra WMMA,
// giving the running softmax denominator replicated in every lane.
// ---------------------------------------------------------------------------
__global__ __launch_bounds__(128)
void attn_flash_kernel(const half_t* __restrict__ kq,
                       const half_t* __restrict__ v,
                       const float*  __restrict__ fr,
                       half_t* __restrict__ ctx) {
    const int bh   = blockIdx.x;
    const int b    = bh / H_;
    const int h    = bh % H_;
    const int q0wg = blockIdx.y * 64;
    const int lane = threadIdx.x & 31;
    const int wave = threadIdx.x >> 5;
    const int q0   = q0wg + wave * 16;
    const int row  = lane & 15;
    const int hl   = lane >> 4;

    __shared__ __attribute__((aligned(32))) half_t vT[64 * 48];      // [dk][k], stride 48
    __shared__ __attribute__((aligned(32))) half_t pL[4][16 * 48];   // per-wave P tile

    // Q fragments: A 16x64 -> two 16x32 fragments
    const half_t* qbase = kq + ((size_t)(b * S_ + q0 + row)) * D_ + h * DK_;
    v16h qa0 = frag_a_from(qbase + 0,  hl);
    v16h qa1 = frag_a_from(qbase + 32, hl);

    // all-ones B fragment (32x16 ones matrix) for row-sum WMMA
    v16h onesb;
    #pragma unroll
    for (int i = 0; i < 16; ++i) onesb[i] = (half_t)1.0f;

    // scale * forget_rate per query row slot (slot j -> row j + 8*hl)
    float frv[8];
    #pragma unroll
    for (int j = 0; j < 8; ++j)
        frv[j] = fr[b * S_ + q0 + j + 8 * hl] * 0.125f;   // 1/sqrt(64)

    float mrun[8];
    #pragma unroll
    for (int j = 0; j < 8; ++j) mrun[j] = -INFINITY;
    v8f o0 = {}, o1 = {}, o2 = {}, o3 = {};
    v8f lsum = {};   // running softmax denominator (every lane holds its rows')

    const int ktend = q0wg + 64 < S_ ? q0wg + 64 : S_;
    for (int kt = 0; kt < ktend; kt += 32) {
        // issue K fragments early so the loads fly across the LDS staging
        const half_t* kb0 = kq + ((size_t)(b * S_ + kt + row)) * D_ + h * DK_;
        const half_t* kb1 = kb0 + (size_t)16 * D_;
        v16h kf00 = *(const v16h*)(kb0 + hl * 16);
        v16h kf01 = *(const v16h*)(kb0 + 32 + hl * 16);
        v16h kf10 = *(const v16h*)(kb1 + hl * 16);
        v16h kf11 = *(const v16h*)(kb1 + 32 + hl * 16);

        // stage V tile (32 keys x 64 dk) transposed into LDS (b128 loads)
        __syncthreads();
        #pragma unroll
        for (int i = threadIdx.x; i < 32 * 8; i += 128) {
            int k = i >> 3, dkb = (i & 7) * 8;
            v8h vv = *(const v8h*)(v + ((size_t)(b * S_ + kt + k)) * D_ + h * DK_ + dkb);
            #pragma unroll
            for (int e = 0; e < 8; ++e) vT[(dkb + e) * 48 + k] = vv[e];
        }
        __syncthreads();

        // scores: two 16-wide key subtiles, contraction over dk=64
        v8f s0 = {}, s1 = {};
        s0 = WMMA_F16(qa0, kf00, s0);
        s0 = WMMA_F16(qa1, kf01, s0);
        s1 = WMMA_F16(qa0, kf10, s1);
        s1 = WMMA_F16(qa1, kf11, s1);

        // scale*fr + strict-lower causal mask
        float sv[2][8];
        v8f sc[2] = {s0, s1};
        #pragma unroll
        for (int sub = 0; sub < 2; ++sub) {
            int tg = kt + sub * 16 + row;
            #pragma unroll
            for (int j = 0; j < 8; ++j) {
                int sg = q0 + j + 8 * hl;
                float xx = sc[sub][j] * frv[j];
                sv[sub][j] = (tg < sg) ? xx : -1.0e32f;
            }
        }

        // online softmax: row max via DPP butterfly (VALU only)
        float alpha[8], mnew[8];
        #pragma unroll
        for (int j = 0; j < 8; ++j) {
            float tm = rowmax16(fmaxf(sv[0][j], sv[1][j]));
            mnew[j]  = fmaxf(mrun[j], tm);
            alpha[j] = __expf(mrun[j] - mnew[j]);
            mrun[j]  = mnew[j];
        }
        #pragma unroll
        for (int j = 0; j < 8; ++j) {
            float p0 = __expf(sv[0][j] - mnew[j]);
            float p1 = __expf(sv[1][j] - mnew[j]);
            int base = (j + 8 * hl) * 48 + row;
            pL[wave][base +  0] = (half_t)p0;
            pL[wave][base + 16] = (half_t)p1;
            o0[j] *= alpha[j]; o1[j] *= alpha[j];
            o2[j] *= alpha[j]; o3[j] *= alpha[j];
            lsum[j] *= alpha[j];
        }

        // ctx += P(16x32) @ V(32x64) ; lsum += P @ ones
        v16h pa  = frag_a_from(&pL[wave][row * 48], hl);
        v16h vb0 = *(const v16h*)(&vT[(0 * 16 + row) * 48 + hl * 16]);
        v16h vb1 = *(const v16h*)(&vT[(1 * 16 + row) * 48 + hl * 16]);
        v16h vb2 = *(const v16h*)(&vT[(2 * 16 + row) * 48 + hl * 16]);
        v16h vb3 = *(const v16h*)(&vT[(3 * 16 + row) * 48 + hl * 16]);
        o0 = WMMA_F16(pa, vb0, o0);
        o1 = WMMA_F16(pa, vb1, o1);
        o2 = WMMA_F16(pa, vb2, o2);
        o3 = WMMA_F16(pa, vb3, o3);
        lsum = WMMA_F16(pa, onesb, lsum);
    }

    // normalize, zero query row 0, store ctx[b, s, h*DK + dk]
    v8f os[4] = {o0, o1, o2, o3};
    #pragma unroll
    for (int j = 0; j < 8; ++j) {
        int sg = q0 + j + 8 * hl;
        float inv = (sg == 0) ? 0.0f : 1.0f / lsum[j];
        half_t* crow = ctx + ((size_t)(b * S_ + sg)) * D_ + h * DK_;
        #pragma unroll
        for (int t = 0; t < 4; ++t)
            crow[t * 16 + row] = (half_t)(os[t][j] * inv);
    }
}

// ---------------------------------------------------------------------------
// x = LayerNorm(x + res) * g + b  ; outputs f32 (in place ok) and f16 shadow
// ---------------------------------------------------------------------------
__global__ __launch_bounds__(256)
void ln_residual_kernel(const float* __restrict__ xin,
                        const float* __restrict__ res,
                        const float* __restrict__ g,
                        const float* __restrict__ beta,
                        float* __restrict__ xout,
                        half_t* __restrict__ xh) {
    const size_t rb = (size_t)blockIdx.x * D_;
    const int t = threadIdx.x;
    float v0 = xin[rb + t]       + res[rb + t];
    float v1 = xin[rb + t + 256] + res[rb + t + 256];
    float s  = v0 + v1;
    float s2 = v0 * v0 + v1 * v1;
    #pragma unroll
    for (int m = 1; m < 32; m <<= 1) {
        s  += __shfl_xor(s,  m, 32);
        s2 += __shfl_xor(s2, m, 32);
    }
    __shared__ float ss[8], ss2[8];
    int wave = t >> 5, lane = t & 31;
    if (lane == 0) { ss[wave] = s; ss2[wave] = s2; }
    __syncthreads();
    float tot = 0.0f, tot2 = 0.0f;
    #pragma unroll
    for (int i = 0; i < 8; ++i) { tot += ss[i]; tot2 += ss2[i]; }
    float mu   = tot  * (1.0f / D_);
    float var  = tot2 * (1.0f / D_) - mu * mu;
    float rstd = rsqrtf(var + EPS_);
    float o0 = (v0 - mu) * rstd * g[t]       + beta[t];
    float o1 = (v1 - mu) * rstd * g[t + 256] + beta[t + 256];
    xout[rb + t]       = o0;  xh[rb + t]       = (half_t)o0;
    xout[rb + t + 256] = o1;  xh[rb + t + 256] = (half_t)o1;
}

// ---------------------------------------------------------------------------
// host side
// ---------------------------------------------------------------------------
extern "C" void kernel_launch(void* const* d_in, const int* in_sizes, int n_in,
                              void* d_out, int out_size, void* d_ws, size_t ws_size,
                              hipStream_t stream) {
    const float* q    = (const float*)d_in[0];
    const float* qa   = (const float*)d_in[1];
    const float* fr   = (const float*)d_in[2];
    const float* pos  = (const float*)d_in[3];
    const float* Wk   = (const float*)d_in[4];
    const float* bk   = (const float*)d_in[5];
    const float* Wv   = (const float*)d_in[6];
    const float* bv   = (const float*)d_in[7];
    const float* Wo   = (const float*)d_in[8];
    const float* bo   = (const float*)d_in[9];
    const float* W1   = (const float*)d_in[10];
    const float* b1   = (const float*)d_in[11];
    const float* W2   = (const float*)d_in[12];
    const float* b2   = (const float*)d_in[13];
    const float* g1   = (const float*)d_in[14];
    const float* be1  = (const float*)d_in[15];
    const float* g2   = (const float*)d_in[16];
    const float* be2  = (const float*)d_in[17];
    float* x = (float*)d_out;                      // running activation (B,S,D) f32

    const size_t nBSD = (size_t)B_ * S_ * D_;      // 8,388,608
    const size_t nBSF = (size_t)B_ * S_ * F_;      // 33,554,432
    const int nDD = L_ * D_ * D_;                  // 1,048,576
    const int nDF = L_ * D_ * F_;                  // 4,194,304

    // workspace layout (~150 MB total)
    char* p = (char*)d_ws;
    half_t* xh   = (half_t*)p; p += nBSD * sizeof(half_t);
    half_t* yh   = (half_t*)p; p += nBSD * sizeof(half_t);
    half_t* big  = (half_t*)p; p += nBSF * sizeof(half_t);  // kq/v/ctx then h1
    float*  tmp  = (float*)p;  p += nBSD * sizeof(float);
    half_t* wkh  = (half_t*)p; p += (size_t)nDD * sizeof(half_t);
    half_t* wvh  = (half_t*)p; p += (size_t)nDD * sizeof(half_t);
    half_t* woh  = (half_t*)p; p += (size_t)nDD * sizeof(half_t);
    half_t* w1h  = (half_t*)p; p += (size_t)nDF * sizeof(half_t);
    half_t* w2h  = (half_t*)p; p += (size_t)nDF * sizeof(half_t);

    half_t* kqh  = big;
    half_t* vh   = big + nBSD;
    half_t* ctxh = big + 2 * nBSD;
    half_t* h1h  = big;                            // reused after attention

    cvt_f16_kernel<<<nDD / 256, 256, 0, stream>>>(Wk, wkh, nDD);
    cvt_f16_kernel<<<nDD / 256, 256, 0, stream>>>(Wv, wvh, nDD);
    cvt_f16_kernel<<<nDD / 256, 256, 0, stream>>>(Wo, woh, nDD);
    cvt_f16_kernel<<<nDF / 256, 256, 0, stream>>>(W1, w1h, nDF);
    cvt_f16_kernel<<<nDF / 256, 256, 0, stream>>>(W2, w2h, nDF);

    add_pos_kernel<<<(int)(nBSD / 256), 256, 0, stream>>>(q, qa, pos, x, xh, yh);

    const int M = B_ * S_;
    dim3 gDD(M / 128, D_ / 128);   // N = 512
    dim3 gDF(M / 128, F_ / 128);   // N = 2048
    dim3 gAttn(B_ * H_, S_ / 64);

    for (int l = 0; l < L_; ++l) {
        const size_t oDD = (size_t)l * D_ * D_;
        const size_t oDF = (size_t)l * D_ * F_;
        // kq = x @ Wk^T + bk ; v = y @ Wv^T + bv   (f16 out)
        gemm_wmma_kernel<false, true><<<gDD, 256, 0, stream>>>(
            xh, wkh + oDD, bk + l * D_, kqh, nullptr, M, D_, D_);
        gemm_wmma_kernel<false, true><<<gDD, 256, 0, stream>>>(
            yh, wvh + oDD, bv + l * D_, vh, nullptr, M, D_, D_);
        // attention
        attn_flash_kernel<<<gAttn, 128, 0, stream>>>(kqh, vh, fr, ctxh);
        // out = ctx @ Wo^T + bo (f32), then x = LN1(x + out)
        gemm_wmma_kernel<false, false><<<gDD, 256, 0, stream>>>(
            ctxh, woh + oDD, bo + l * D_, nullptr, tmp, M, D_, D_);
        ln_residual_kernel<<<M, 256, 0, stream>>>(x, tmp, g1 + l * D_, be1 + l * D_, x, xh);
        // h1 = relu(x @ W1^T + b1) (f16) ; ff = h1 @ W2^T + b2 (f32)
        gemm_wmma_kernel<true, true><<<gDF, 256, 0, stream>>>(
            xh, w1h + oDF, b1 + l * F_, h1h, nullptr, M, F_, D_);
        gemm_wmma_kernel<false, false><<<gDD, 256, 0, stream>>>(
            h1h, w2h + oDF, b2 + l * D_, nullptr, tmp, M, D_, F_);
        ln_residual_kernel<<<M, 256, 0, stream>>>(x, tmp, g2 + l * D_, be2 + l * D_, x, xh);
    }
}
